// TransforemrLayers_LS_24026047054601
// MI455X (gfx1250) — matrix operand order repaired and checked by reading, not compiled, not run on hardware
//
#include <hip/hip_runtime.h>
#include <hip/hip_bf16.h>
#include <cmath>

// ---------------- types / helpers ----------------
typedef _Float16 v8h  __attribute__((ext_vector_type(8)));
typedef _Float16 v16h __attribute__((ext_vector_type(16)));
typedef float    v8f  __attribute__((ext_vector_type(8)));

#define WMMA_F32_F16(a, b, c) \
  __builtin_amdgcn_wmma_f32_16x16x32_f16(false, (a), false, (b), (short)0, (c), false, false)

static constexpr int Lc = 2, Dc = 512, Hc = 8, DKc = 64, Rc = 64, Wc = 64, DFFc = 2048;
static constexpr int HDc = 512, HRc = 512, Bc = 4, Sc = 4096, Mc = Bc * Sc; // 16384

// ---------------- CDNA5 Tensor Data Mover (async tile DMA global->LDS) ----------------
#if __has_builtin(__builtin_amdgcn_tensor_load_to_lds)
#define HAVE_TDM 1
typedef unsigned int u32x4 __attribute__((ext_vector_type(4)));
typedef int          i32x8 __attribute__((ext_vector_type(8)));
typedef int          i32x4 __attribute__((ext_vector_type(4)));
#endif

#if __has_builtin(__builtin_amdgcn_s_wait_tensorcnt)
#define TENSOR_WAITN(n) __builtin_amdgcn_s_wait_tensorcnt((short)(n))
#else
#define TENSOR_WAITN(n) asm volatile("s_wait_tensorcnt " #n ::: "memory")
#endif

__device__ __forceinline__ unsigned lds_addr_of(const void* p) {
  return (unsigned)(unsigned long long)(uintptr_t)p; // LDS aperture: low 32 bits = LDS byte offset
}

#ifdef HAVE_TDM
// 2D tile DMA: rows x row_elems f16 elements, source row stride src_stride elements,
// LDS destination padded by pad_amt code after every pad_int code DWORDs (ISA D# encoding).
__device__ __forceinline__ void tdm_load_tile(unsigned lds_addr, const _Float16* g,
                                              unsigned rows, unsigned row_elems,
                                              unsigned src_stride,
                                              unsigned pad_int, unsigned pad_amt) {
  const unsigned long long ga = (unsigned long long)(uintptr_t)g;
  u32x4 g0;
  g0[0] = 1u;                                   // count=1 (valid user descriptor)
  g0[1] = lds_addr;                             // lds_addr [63:32]
  g0[2] = (unsigned)ga;                         // global_addr low  [95:64]
  g0[3] = (unsigned)(ga >> 32) | (2u << 30);    // global_addr high [120:96] | type=2
  i32x8 g1;
  g1[0] = (int)((1u << 16) | (1u << 20) | (pad_int << 22) | (pad_amt << 25)); // data_size=2B, pad_enable
  g1[1] = (int)((src_stride & 0xFFFFu) << 16);                // tensor_dim0[15:0]  @ [63:48]
  g1[2] = (int)((src_stride >> 16) | (rows << 16));           // tensor_dim0[31:16] | tensor_dim1[15:0]
  g1[3] = (int)((rows >> 16) | (row_elems << 16));            // tensor_dim1[31:16] | tile_dim0
  g1[4] = (int)rows;                                          // tile_dim1 | tile_dim2=0
  g1[5] = (int)src_stride;                                    // tensor_dim0_stride low32
  g1[6] = 0;                                                  // stride hi | dim1_stride lo
  g1[7] = 0;
  const i32x4 z4 = {0, 0, 0, 0};
  const i32x8 z8 = {0, 0, 0, 0, 0, 0, 0, 0};
  __builtin_amdgcn_tensor_load_to_lds(g0, g1, z4, z4, z8, 0);
}
#endif

// Build a 16-half fragment from two 16B-aligned 8-half LDS chunks.
__device__ __forceinline__ v16h ld16p(const _Float16* p0, const _Float16* p1) {
  v8h a = *(const v8h*)p0;
  v8h b = *(const v8h*)p1;
  v16h r;
#pragma unroll
  for (int i = 0; i < 8; ++i) { r[i] = a[i]; r[i + 8] = b[i]; }
  return r;
}

// ---------------- weight convert + transpose: dst[n*K+k] = (f16)(src[k*N+n]*scale) ----------------
__global__ __launch_bounds__(256) void convT_kernel(const float* __restrict__ src,
                                                    _Float16* __restrict__ dst,
                                                    int K, int N, float scale) {
  const int idx = blockIdx.x * 256 + threadIdx.x;
  const int n = idx / K, k = idx % K;
  dst[(size_t)n * K + k] = (_Float16)(src[(size_t)k * N + n] * scale);
}

// ---------------- LayerNorm (row length 512), f32 in -> f16 out ----------------
__global__ __launch_bounds__(256) void ln_kernel(const float* __restrict__ in,
                                                 _Float16* __restrict__ out,
                                                 const float* __restrict__ g,
                                                 const float* __restrict__ bta) {
  const int row = blockIdx.x;
  const int t = threadIdx.x;
  const float* ip = in + (size_t)row * 512;
  float x0 = ip[t], x1 = ip[t + 256];
  __shared__ float red[256];
  red[t] = x0 + x1;
  __syncthreads();
  for (int off = 128; off > 0; off >>= 1) {
    if (t < off) red[t] += red[t + off];
    __syncthreads();
  }
  const float mean = red[0] * (1.0f / 512.0f);
  __syncthreads();
  const float d0 = x0 - mean, d1 = x1 - mean;
  red[t] = d0 * d0 + d1 * d1;
  __syncthreads();
  for (int off = 128; off > 0; off >>= 1) {
    if (t < off) red[t] += red[t + off];
    __syncthreads();
  }
  const float rstd = rsqrtf(red[0] * (1.0f / 512.0f) + 1e-5f);
  _Float16* op = out + (size_t)row * 512;
  op[t]       = (_Float16)(d0 * rstd * g[t]       + bta[t]);
  op[t + 256] = (_Float16)(d1 * rstd * g[t + 256] + bta[t + 256]);
}

// ---------------- WMMA GEMM: C(MxN) = A(MxK,f16) * Bt(NxK,f16)^T + bias*bscale [+gelu][+res] ----
// block tile 128x128, 256 threads (8 waves), each wave 32x64, K step 32.
// Double-buffered LDS; TDM pipeline: issue stage i+1, s_wait_tensorcnt 2 => stage i complete.
template <int ACT, bool RES, bool OUTH>
__global__ __launch_bounds__(256) void gemm_kernel(const _Float16* __restrict__ A,
                                                   const _Float16* __restrict__ Bt,
                                                   const float* __restrict__ bias, float bscale,
                                                   const float* Res, void* C, int K, int N) {
  __shared__ _Float16 As[2][128 * 40]; // [m][k] stride 40 halves (64B data + 16B pad)
  __shared__ _Float16 Bs[2][128 * 40]; // [n][k]
  const int tid = threadIdx.x;
  const int lane = tid & 31, wv = tid >> 5;
  const int ln15 = lane & 15, hi = lane >> 4, akb = hi * 16;
  const int wm = (wv & 3) * 32, wn = (wv >> 2) * 64;
  const int m0 = blockIdx.y * 128, n0 = blockIdx.x * 128;
  const int nk = K >> 5;
  v8f acc[2][4] = {};
#ifdef HAVE_TDM
  const unsigned lAs[2] = {lds_addr_of(As[0]), lds_addr_of(As[1])};
  const unsigned lBs[2] = {lds_addr_of(Bs[0]), lds_addr_of(Bs[1])};
  if (wv == 0) { // prologue: stage 0
    tdm_load_tile(lAs[0], A + (size_t)m0 * K, 128, 32, (unsigned)K, 3, 3); // 16DW rows, +4DW pad
    tdm_load_tile(lBs[0], Bt + (size_t)n0 * K, 128, 32, (unsigned)K, 3, 3);
  }
#endif

  for (int it = 0; it < nk; ++it) {
    const int buf = it & 1;
#ifdef HAVE_TDM
    if (wv == 0) {
      if (it + 1 < nk) { // issue next stage into alternate buffer, then wait for current stage
        const int kt1 = (it + 1) << 5;
        tdm_load_tile(lAs[buf ^ 1], A + (size_t)m0 * K + kt1, 128, 32, (unsigned)K, 3, 3);
        tdm_load_tile(lBs[buf ^ 1], Bt + (size_t)n0 * K + kt1, 128, 32, (unsigned)K, 3, 3);
        TENSOR_WAITN(2); // in-order: oldest 2 (stage it) retired, next 2 still in flight
      } else {
        TENSOR_WAITN(0);
      }
    }
#else
    {
      const int kt = it << 5;
      for (int i = tid; i < 512; i += 256) {
        const int r = i >> 2, ch = (i & 3) << 3;
        *(v8h*)&As[buf][r * 40 + ch] = *(const v8h*)(A + (size_t)(m0 + r) * K + kt + ch);
      }
      for (int i = tid; i < 512; i += 256) {
        const int r = i >> 2, ch = (i & 3) << 3;
        *(v8h*)&Bs[buf][r * 40 + ch] = *(const v8h*)(Bt + (size_t)(n0 + r) * K + kt + ch);
      }
    }
#endif
    __syncthreads(); // stage `it` visible to all waves
    v16h af[2], bf[4];
#pragma unroll
    for (int mt = 0; mt < 2; ++mt) {
      const _Float16* p = &As[buf][(wm + mt * 16 + ln15) * 40 + akb];
      af[mt] = ld16p(p, p + 8);
    }
#pragma unroll
    for (int nt = 0; nt < 4; ++nt) {
      const _Float16* p = &Bs[buf][(wn + nt * 16 + ln15) * 40 + hi * 16];
      bf[nt] = ld16p(p, p + 8);
    }
#pragma unroll
    for (int mt = 0; mt < 2; ++mt)
#pragma unroll
      for (int nt = 0; nt < 4; ++nt)
        acc[mt][nt] = WMMA_F32_F16(af[mt], bf[nt], acc[mt][nt]);
    __syncthreads(); // all waves done with buf before stage it+2 overwrites it
  }
  // epilogue
#pragma unroll
  for (int mt = 0; mt < 2; ++mt)
#pragma unroll
    for (int nt = 0; nt < 4; ++nt) {
      const int col = n0 + wn + nt * 16 + ln15;
      const float bsv = bias[col] * bscale;
#pragma unroll
      for (int r = 0; r < 8; ++r) {
        const int row = m0 + wm + mt * 16 + hi * 8 + r;
        float v = acc[mt][nt][r] + bsv;
        if (ACT == 1) v = 0.5f * v * (1.0f + erff(v * 0.70710678118654752f));
        const size_t o = (size_t)row * N + col;
        if (OUTH) {
          ((_Float16*)C)[o] = (_Float16)v;
        } else {
          if (RES) v += Res[o];
          ((float*)C)[o] = v;
        }
      }
    }
}

// ---------------- hs softmax over axis=1 (S), masked; f32 in, f16 out ----------------
__global__ __launch_bounds__(256) void hs_part_kernel(const float* __restrict__ hs,
                                                      const int* __restrict__ mask,
                                                      float* __restrict__ pM,
                                                      float* __restrict__ pL) {
  const int ch = blockIdx.x, b = blockIdx.y;
  const int s0 = ch * 256;
  for (int cc = 0; cc < 2; ++cc) {
    const int col = threadIdx.x + cc * 256;
    float m = -INFINITY, l = 0.f;
    for (int r = 0; r < 256; ++r) {
      const int s = s0 + r;
      if (mask[b * Sc + s] != 0) {
        const float x = hs[(size_t)(b * Sc + s) * HRc + col];
        const float nm = fmaxf(m, x);
        l = l * __expf(m - nm) + __expf(x - nm);
        m = nm;
      }
    }
    pM[((size_t)b * 16 + ch) * HRc + col] = m;
    pL[((size_t)b * 16 + ch) * HRc + col] = l;
  }
}

__global__ __launch_bounds__(256) void hs_norm_kernel(const float* __restrict__ hs,
                                                      _Float16* __restrict__ out,
                                                      const int* __restrict__ mask,
                                                      const float* __restrict__ pM,
                                                      const float* __restrict__ pL) {
  const int ch = blockIdx.x, b = blockIdx.y;
  const int s0 = ch * 256;
  for (int cc = 0; cc < 2; ++cc) {
    const int col = threadIdx.x + cc * 256;
    float gm = -INFINITY, gl = 0.f;
    for (int k = 0; k < 16; ++k) {
      const float pm = pM[((size_t)b * 16 + k) * HRc + col];
      const float pl = pL[((size_t)b * 16 + k) * HRc + col];
      if (pl > 0.f) {
        const float nm = fmaxf(gm, pm);
        gl = gl * __expf(gm - nm) + pl * __expf(pm - nm);
        gm = nm;
      }
    }
    const float inv = (gl > 0.f) ? 1.f / gl : 0.f;
    for (int r = 0; r < 256; ++r) {
      const int s = s0 + r;
      const size_t o = (size_t)(b * Sc + s) * HRc + col;
      out[o] = (_Float16)((mask[b * Sc + s] != 0) ? __expf(hs[o] - gm) * inv : 0.f);
    }
  }
}

// ---------------- landmark compression: kc/vc[b,h,r,d] = sum_s hs[b,s,h,r]*k/v[b,s,h,d] ----------
__global__ __launch_bounds__(256) void compress_kernel(const _Float16* __restrict__ hsh,
                                                       const _Float16* __restrict__ kh,
                                                       const _Float16* __restrict__ vh,
                                                       float* __restrict__ kc,
                                                       float* __restrict__ vcc) {
  __shared__ _Float16 Hs[64 * 40]; // [r][s32] (transposed on stage)
  __shared__ _Float16 Ks[64 * 40]; // [d][s32]
  __shared__ _Float16 Vs[64 * 40];
  const int b = blockIdx.x >> 3, h = blockIdx.x & 7;
  const int tid = threadIdx.x, lane = tid & 31, wv = tid >> 5;
  const int ln15 = lane & 15, hi = lane >> 4, akb = hi * 16;
  const int mt = wv & 3;
  v8f acc[4] = {};
  for (int st = 0; st < Sc; st += 32) {
    {  // 256 threads x (1 v8h load + 8 b16 LDS stores) per array
      const int s = tid >> 3, r8 = (tid & 7) << 3;
      const size_t g = (size_t)(b * Sc + st + s) * 512 + h * 64 + r8;
      const v8h hv = *(const v8h*)(hsh + g);
      const v8h kv = *(const v8h*)(kh + g);
      const v8h vv = *(const v8h*)(vh + g);
#pragma unroll
      for (int j = 0; j < 8; ++j) {
        Hs[(r8 + j) * 40 + s] = hv[j];
        Ks[(r8 + j) * 40 + s] = kv[j];
        Vs[(r8 + j) * 40 + s] = vv[j];
      }
    }
    __syncthreads();
    const _Float16* ap = &Hs[(mt * 16 + ln15) * 40 + akb];
    const v16h a = ld16p(ap, ap + 8);
    const _Float16* base = (wv < 4) ? Ks : Vs;
#pragma unroll
    for (int nt = 0; nt < 4; ++nt) {
      const _Float16* bp = &base[(nt * 16 + ln15) * 40 + hi * 16];
      acc[nt] = WMMA_F32_F16(a, ld16p(bp, bp + 8), acc[nt]);
    }
    __syncthreads();
  }
  float* outp = (wv < 4) ? kc : vcc;
#pragma unroll
  for (int nt = 0; nt < 4; ++nt)
#pragma unroll
    for (int r = 0; r < 8; ++r) {
      const int rr = mt * 16 + hi * 8 + r;
      const int dd = nt * 16 + ln15;
      outp[(size_t)(b * Rc + rr) * HDc + h * 64 + dd] = acc[nt][r];
    }
}

// ---------------- fused LS attention (f16 in, f16 out) ----------------
__global__ __launch_bounds__(256) void attn_kernel(const _Float16* __restrict__ qh,
                                                   const _Float16* __restrict__ kh,
                                                   const _Float16* __restrict__ vh,
                                                   const _Float16* __restrict__ kch,
                                                   const _Float16* __restrict__ vch,
                                                   const int* __restrict__ mask,
                                                   _Float16* __restrict__ outc) {
  __shared__ __align__(16) char smem[13824 + 18432 + 25088]; // 57,344 B
  _Float16* Qs   = (_Float16*)smem;                // [32][72]
  _Float16* KCs  = (_Float16*)(smem + 4608);       // [64][72]
  _Float16* KTs  = (_Float16*)(smem + 13824);      // [128][72]
  _Float16* Pb   = (_Float16*)smem;                // [32][200] overlay (Q/KC dead)
  _Float16* Vbuf = (_Float16*)(smem + 13824);      // overlay (KT dead): [64][72] then [64][136]
  float*    Ss   = (float*)(smem + 13824 + 18432); // [32][196] f32 scores

  const int c = blockIdx.x >> 1, qhf = blockIdx.x & 1;
  const int h = blockIdx.y, b = blockIdx.z;
  const int qbase = c * 64 + qhf * 32;
  const int ws = (c < 62 ? c : 62) * 64; // start = min(c*W, S-2W)
  const int tid = threadIdx.x, lane = tid & 31, wv = tid >> 5;
  const int ln15 = lane & 15, hi = lane >> 4, akb = hi * 16;

  // stage Q (scale pre-folded into wq/bq), compressed-K, window-K
#ifdef HAVE_TDM
  if (wv == 0) {
    tdm_load_tile(lds_addr_of(Qs),  qh  + (size_t)(b * Sc + qbase) * 512 + h * 64, 32, 64, 512, 4, 3);
    tdm_load_tile(lds_addr_of(KCs), kch + (size_t)(b * Rc) * 512 + h * 64,          64, 64, 512, 4, 3);
    tdm_load_tile(lds_addr_of(KTs), kh  + (size_t)(b * Sc + ws) * 512 + h * 64,    128, 64, 512, 4, 3);
    TENSOR_WAITN(0);
  }
#else
  {
    const int r = tid >> 3, ch2 = (tid & 7) << 3;
    *(v8h*)&Qs[r * 72 + ch2] = *(const v8h*)(qh + (size_t)(b * Sc + qbase + r) * 512 + h * 64 + ch2);
  }
  for (int i = tid; i < 512; i += 256) {
    const int r = i >> 3, ch2 = (i & 7) << 3;
    *(v8h*)&KCs[r * 72 + ch2] = *(const v8h*)(kch + (size_t)(b * Rc + r) * 512 + h * 64 + ch2);
  }
  for (int i = tid; i < 1024; i += 256) {
    const int r = i >> 3, ch2 = (i & 7) << 3;
    *(v8h*)&KTs[r * 72 + ch2] = *(const v8h*)(kh + (size_t)(b * Sc + ws + r) * 512 + h * 64 + ch2);
  }
#endif
  __syncthreads();

  // scores: 32x192 = 2 M-tiles x 12 N-tiles, 8 waves
  {
    const int mt = wv & 1, g = wv >> 1;
    for (int nt = g * 3; nt < g * 3 + 3; ++nt) {
      v8f acc = {};
#pragma unroll
      for (int ks = 0; ks < 2; ++ks) {
        const _Float16* ap = &Qs[(mt * 16 + ln15) * 72 + ks * 32 + akb];
        const v16h a = ld16p(ap, ap + 8);
        const _Float16* bp = (nt < 4)
            ? &KCs[(nt * 16 + ln15) * 72 + ks * 32 + hi * 16]
            : &KTs[((nt - 4) * 16 + ln15) * 72 + ks * 32 + hi * 16];
        acc = WMMA_F32_F16(a, ld16p(bp, bp + 8), acc);
      }
#pragma unroll
      for (int r = 0; r < 8; ++r)
        Ss[(mt * 16 + hi * 8 + r) * 196 + nt * 16 + ln15] = acc[r];
    }
  }
  __syncthreads();

  // masked softmax over 192; query-mask zeroes row
  if (tid < 32) {
    const int qm = mask[b * Sc + qbase + tid];
    float m = -INFINITY;
    for (int j = 0; j < 192; ++j) {
      const bool valid = (j < 64) || (mask[b * Sc + ws + (j - 64)] != 0);
      if (valid) m = fmaxf(m, Ss[tid * 196 + j]);
    }
    float l = 0.f;
    for (int j = 0; j < 192; ++j) {
      const bool valid = (j < 64) || (mask[b * Sc + ws + (j - 64)] != 0);
      if (valid) l += __expf(Ss[tid * 196 + j] - m);
    }
    const float inv = (qm != 0 && l > 0.f) ? 1.f / l : 0.f;
    for (int j = 0; j < 192; ++j) {
      const bool valid = (j < 64) || (mask[b * Sc + ws + (j - 64)] != 0);
      Pb[tid * 200 + j] = (_Float16)(valid ? __expf(Ss[tid * 196 + j] - m) * inv : 0.f);
    }
  }
  __syncthreads();

  // O = P[:, :64] @ VC (stage VC^T into dead KT region)
  const int mt = wv & 1, nt = wv >> 1;
  v8f oacc = {};
  for (int i = tid; i < 512; i += 256) {
    const int r = i >> 3, d8 = (i & 7) << 3;
    const v8h vv = *(const v8h*)(vch + (size_t)(b * Rc + r) * 512 + h * 64 + d8);
#pragma unroll
    for (int j = 0; j < 8; ++j) Vbuf[(d8 + j) * 72 + r] = vv[j];
  }
  __syncthreads();
#pragma unroll
  for (int ks = 0; ks < 2; ++ks) {
    const _Float16* ap = &Pb[(mt * 16 + ln15) * 200 + ks * 32 + akb];
    const v16h a = ld16p(ap, ap + 8);
    const _Float16* bp = &Vbuf[(nt * 16 + ln15) * 72 + ks * 32 + hi * 16];
    oacc = WMMA_F32_F16(a, ld16p(bp, bp + 8), oacc);
  }
  __syncthreads();

  // O += P[:, 64:192] @ V_window (restage window-V^T)
  for (int i = tid; i < 1024; i += 256) {
    const int k = i >> 3, d8 = (i & 7) << 3;
    const v8h vv = *(const v8h*)(vh + (size_t)(b * Sc + ws + k) * 512 + h * 64 + d8);
#pragma unroll
    for (int j = 0; j < 8; ++j) Vbuf[(d8 + j) * 136 + k] = vv[j];
  }
  __syncthreads();
#pragma unroll
  for (int ks = 0; ks < 4; ++ks) {
    const _Float16* ap = &Pb[(mt * 16 + ln15) * 200 + 64 + ks * 32 + akb];
    const v16h a = ld16p(ap, ap + 8);
    const _Float16* bp = &Vbuf[(nt * 16 + ln15) * 136 + ks * 32 + hi * 16];
    oacc = WMMA_F32_F16(a, ld16p(bp, bp + 8), oacc);
  }
#pragma unroll
  for (int r = 0; r < 8; ++r) {
    const int row = qbase + mt * 16 + hi * 8 + r;
    const int col = h * 64 + nt * 16 + ln15;
    outc[(size_t)(b * Sc + row) * HDc + col] = (_Float16)oacc[r];
  }
}

// ---------------- host ----------------
static void launch_gemm(hipStream_t s, const _Float16* A, const _Float16* Bt, const float* bias,
                        float bscale, const float* Res, void* C, int K, int N, int act, int outh) {
  dim3 g(N / 128, Mc / 128);
  if (act == 1)
    gemm_kernel<1, false, true><<<g, 256, 0, s>>>(A, Bt, bias, bscale, nullptr, C, K, N);
  else if (outh)
    gemm_kernel<0, false, true><<<g, 256, 0, s>>>(A, Bt, bias, bscale, nullptr, C, K, N);
  else if (Res)
    gemm_kernel<0, true, false><<<g, 256, 0, s>>>(A, Bt, bias, bscale, Res, C, K, N);
  else
    gemm_kernel<0, false, false><<<g, 256, 0, s>>>(A, Bt, bias, bscale, nullptr, C, K, N);
}

extern "C" void kernel_launch(void* const* d_in, const int* in_sizes, int n_in,
                              void* d_out, int out_size, void* d_ws, size_t ws_size,
                              hipStream_t stream) {
  (void)in_sizes; (void)n_in; (void)out_size; (void)ws_size;
  const float* x     = (const float*)d_in[0];
  const int*   mask  = (const int*)d_in[1];
  const float* ln1_g = (const float*)d_in[2];
  const float* ln1_b = (const float*)d_in[3];
  const float* wq    = (const float*)d_in[4];
  const float* bq    = (const float*)d_in[5];
  const float* wk    = (const float*)d_in[6];
  const float* bk    = (const float*)d_in[7];
  const float* wvw   = (const float*)d_in[8];
  const float* bv    = (const float*)d_in[9];
  const float* wo    = (const float*)d_in[10];
  const float* bo    = (const float*)d_in[11];
  const float* lns_g = (const float*)d_in[12];
  const float* lns_b = (const float*)d_in[13];
  const float* lnl_g = (const float*)d_in[14];
  const float* lnl_b = (const float*)d_in[15];
  const float* dw    = (const float*)d_in[16];
  const float* db    = (const float*)d_in[17];
  const float* ln2_g = (const float*)d_in[18];
  const float* ln2_b = (const float*)d_in[19];
  const float* fc1w  = (const float*)d_in[20];
  const float* fc1b  = (const float*)d_in[21];
  const float* fc2w  = (const float*)d_in[22];
  const float* fc2b  = (const float*)d_in[23];

  char* wsp = (char*)d_ws;
  auto take = [&](size_t bytes) -> void* {
    void* p = (void*)wsp;
    wsp += (bytes + 255) & ~(size_t)255;
    return p;
  };
  float*    xcur = (float*)take((size_t)Mc * Dc * 4);
  float*    tmp  = (float*)take((size_t)Mc * 512 * 4);  // k/v/hs f32 staging
  _Float16* xlnh = (_Float16*)take((size_t)Mc * 512 * 2);
  _Float16* qh   = (_Float16*)take((size_t)Mc * 512 * 2);
  _Float16* kh   = (_Float16*)take((size_t)Mc * 512 * 2);
  _Float16* vh   = (_Float16*)take((size_t)Mc * 512 * 2);
  _Float16* hsh  = (_Float16*)take((size_t)Mc * 512 * 2); // hs f16; later attention output
  _Float16* f1h  = (_Float16*)take((size_t)Mc * DFFc * 2);
  float*    pM   = (float*)take((size_t)Bc * 16 * HRc * 4);
  float*    pL   = (float*)take((size_t)Bc * 16 * HRc * 4);
  float*    kcb  = (float*)take((size_t)Bc * Rc * HDc * 4);
  float*    vcb  = (float*)take((size_t)Bc * Rc * HDc * 4);
  _Float16* kch  = (_Float16*)take((size_t)Bc * Rc * HDc * 2);
  _Float16* vch  = (_Float16*)take((size_t)Bc * Rc * HDc * 2);
  _Float16* wqT  = (_Float16*)take((size_t)Dc * HDc * 2);
  _Float16* wkT  = (_Float16*)take((size_t)Dc * HDc * 2);
  _Float16* wvT  = (_Float16*)take((size_t)Dc * HDc * 2);
  _Float16* dwT  = (_Float16*)take((size_t)Dc * HRc * 2);
  _Float16* woT  = (_Float16*)take((size_t)HDc * Dc * 2);
  _Float16* fc1T = (_Float16*)take((size_t)Dc * DFFc * 2);
  _Float16* fc2T = (_Float16*)take((size_t)DFFc * Dc * 2);

  for (int l = 0; l < Lc; ++l) {
    const float* xin = l ? xcur : x;
    // weight convert + transpose (q-scale folded into wq)
    convT_kernel<<<Dc * HDc / 256, 256, 0, stream>>>(wq + (size_t)l * Dc * HDc, wqT, Dc, HDc, 0.125f);
    convT_kernel<<<Dc * HDc / 256, 256, 0, stream>>>(wk + (size_t)l * Dc * HDc, wkT, Dc, HDc, 1.0f);
    convT_kernel<<<Dc * HDc / 256, 256, 0, stream>>>(wvw + (size_t)l * Dc * HDc, wvT, Dc, HDc, 1.0f);
    convT_kernel<<<Dc * HRc / 256, 256, 0, stream>>>(dw + (size_t)l * Dc * HRc, dwT, Dc, HRc, 1.0f);
    convT_kernel<<<HDc * Dc / 256, 256, 0, stream>>>(wo + (size_t)l * HDc * Dc, woT, HDc, Dc, 1.0f);
    convT_kernel<<<Dc * DFFc / 256, 256, 0, stream>>>(fc1w + (size_t)l * Dc * DFFc, fc1T, Dc, DFFc, 1.0f);
    convT_kernel<<<DFFc * Dc / 256, 256, 0, stream>>>(fc2w + (size_t)l * DFFc * Dc, fc2T, DFFc, Dc, 1.0f);

    // ln1 + projections
    ln_kernel<<<Mc, 256, 0, stream>>>(xin, xlnh, ln1_g + l * Dc, ln1_b + l * Dc);
    launch_gemm(stream, xlnh, wqT, bq + l * HDc, 0.125f, nullptr, qh, Dc, HDc, 0, 1);
    launch_gemm(stream, xlnh, wkT, bk + l * HDc, 1.0f, nullptr, tmp, Dc, HDc, 0, 0);
    ln_kernel<<<Mc, 256, 0, stream>>>(tmp, kh, lns_g + l * HDc, lns_b + l * HDc);
    launch_gemm(stream, xlnh, wvT, bv + l * HDc, 1.0f, nullptr, tmp, Dc, HDc, 0, 0);
    ln_kernel<<<Mc, 256, 0, stream>>>(tmp, vh, lns_g + l * HDc, lns_b + l * HDc);
    launch_gemm(stream, xlnh, dwT, db + l * HRc, 1.0f, nullptr, tmp, Dc, HRc, 0, 0);
    // hs softmax over S
    hs_part_kernel<<<dim3(16, Bc), 256, 0, stream>>>(tmp, mask, pM, pL);
    hs_norm_kernel<<<dim3(16, Bc), 256, 0, stream>>>(tmp, hsh, mask, pM, pL);
    // landmark compression + ln
    compress_kernel<<<Bc * Hc, 256, 0, stream>>>(hsh, kh, vh, kcb, vcb);
    ln_kernel<<<Bc * Rc, 256, 0, stream>>>(kcb, kch, lnl_g + l * HDc, lnl_b + l * HDc);
    ln_kernel<<<Bc * Rc, 256, 0, stream>>>(vcb, vch, lnl_g + l * HDc, lnl_b + l * HDc);
    // fused LS attention (writes into hsh, dead as hs now)
    attn_kernel<<<dim3((Sc / Wc) * 2, Hc, Bc), 256, 0, stream>>>(qh, kh, vh, kch, vch, mask, hsh);
    // out projection + residual
    launch_gemm(stream, hsh, woT, bo + l * Dc, 1.0f, xin, xcur, HDc, Dc, 0, 0);
    // FFN
    ln_kernel<<<Mc, 256, 0, stream>>>(xcur, xlnh, ln2_g + l * Dc, ln2_b + l * Dc);
    launch_gemm(stream, xlnh, fc1T, fc1b + l * DFFc, 1.0f, nullptr, f1h, Dc, DFFc, 1, 1);
    float* outp = (l == Lc - 1) ? (float*)d_out : xcur;
    launch_gemm(stream, f1h, fc2T, fc2b + l * Dc, 1.0f, xcur, outp, DFFc, Dc, 0, 0);
  }
}
